// DeepReasoningGNN_89189290869066
// MI455X (gfx1250) — compile-verified, matrix operand order checked by reading, not run on hardware
//
#include <hip/hip_runtime.h>
#include <hip/hip_bf16.h>

#define N_NODES 50000
#define N_PAD   50048          // 391 * 128: padded row count for workspace buffers
#define N_EDGES 640000
#define NUM_GRAPHS 64
#define HID 128
#define NUM_LAYERS 4

typedef __attribute__((ext_vector_type(16))) __bf16 bf16x16;
typedef __attribute__((ext_vector_type(8)))  float  floatx8;

union Frag {
    bf16x16 v;
    uint4 q[2];
};

#define LDS_PAD 8
#define LDS_STRIDE (HID + LDS_PAD)

__device__ __forceinline__ unsigned pack_bf16(float a, float b) {
    union { __hip_bfloat16 h; unsigned short u; } ua, ub;
    ua.h = __float2bfloat16(a);
    ub.h = __float2bfloat16(b);
    return (unsigned)ua.u | ((unsigned)ub.u << 16);
}

// ---------------------------------------------------------------------------
// C[M x 128] = act( A[M x 128] @ W[128 x 128] + bias ), bf16 WMMA, f32 accum.
// Wt is the PRE-TRANSPOSED bf16 weight: Wt[n][k]. Optionally also writes
// AGG[r][n] = C[r][n] * dinv[r]^2 (fused GCN self-loop init).
// ---------------------------------------------------------------------------
template<bool RELU, bool BIAS, bool GUARD_STORE, bool AGG_OUT>
__global__ __launch_bounds__(256)
void gemm128_wmma(const float* __restrict__ A,
                  const __hip_bfloat16* __restrict__ Wt,
                  const float* __restrict__ bias,
                  float* __restrict__ C,
                  float* __restrict__ AGG,
                  const float* __restrict__ dinv,
                  int Mload, int Mstore)
{
    __shared__ __align__(16) __hip_bfloat16 sA[128 * LDS_STRIDE];  // [row][k]
    __shared__ __align__(16) __hip_bfloat16 sW[128 * LDS_STRIDE];  // [n][k]

    const int tid  = threadIdx.x;
    const int row0 = blockIdx.x * 128;

    // Stage A tile: float4 global loads -> packed bf16 ds_store_b64
    for (int i = tid; i < 128 * 32; i += 256) {
        int r = i >> 5, c4 = (i & 31) << 2;
        float4 v = make_float4(0.f, 0.f, 0.f, 0.f);
        if (row0 + r < Mload) v = *(const float4*)(A + (size_t)(row0 + r) * HID + c4);
        uint2 p;
        p.x = pack_bf16(v.x, v.y);
        p.y = pack_bf16(v.z, v.w);
        *(uint2*)(&sA[r * LDS_STRIDE + c4]) = p;
    }
    // Stage W (already bf16 + transposed): straight b128 copies
    for (int i = tid; i < 128 * 16; i += 256) {
        int nrow = i >> 4, k8 = (i & 15) << 3;
        *(uint4*)(&sW[nrow * LDS_STRIDE + k8]) = *(const uint4*)(Wt + (size_t)nrow * HID + k8);
    }
    __syncthreads();

    const int wave = tid >> 5;
    const int lane = tid & 31;
    const int half = lane >> 4;
    const int l15  = lane & 15;
    const int mrow = wave * 16;

    // Preload the 4 A fragments (K steps of 32): two b128 LDS reads each
    Frag afr[4];
#pragma unroll
    for (int ks = 0; ks < 4; ++ks) {
        const __hip_bfloat16* p = &sA[(mrow + l15) * LDS_STRIDE + ks * 32 + half * 8];
        afr[ks].q[0] = *(const uint4*)(p);
        afr[ks].q[1] = *(const uint4*)(p + 16);
    }

    // Per-row dinv^2 for fused AGG init (same rows for all 8 n-tiles)
    float dv2[8];
    if (AGG_OUT) {
#pragma unroll
        for (int r = 0; r < 8; ++r) {
            float dv = dinv[row0 + mrow + half * 8 + r];
            dv2[r] = dv * dv;
        }
    }

#pragma unroll
    for (int nt = 0; nt < 8; ++nt) {
        floatx8 acc = {};
#pragma unroll
        for (int ks = 0; ks < 4; ++ks) {
            Frag bfr;
            const __hip_bfloat16* p = &sW[(nt * 16 + l15) * LDS_STRIDE + ks * 32 + half * 8];
            bfr.q[0] = *(const uint4*)(p);
            bfr.q[1] = *(const uint4*)(p + 16);
            acc = __builtin_amdgcn_wmma_f32_16x16x32_bf16(
                false, afr[ks].v, false, bfr.v, (short)0, acc, false, false);
        }
        const int n = nt * 16 + l15;
        float bv = 0.0f;
        if (BIAS) bv = bias[n];
#pragma unroll
        for (int r = 0; r < 8; ++r) {
            const int gr = row0 + mrow + half * 8 + r;  // C layout: VGPR r -> M = r + 8*half
            float v = acc[r];
            if (BIAS) v += bv;
            if (RELU) v = fmaxf(v, 0.0f);
            if (!GUARD_STORE || gr < Mstore) {
                C[(size_t)gr * HID + n] = v;
                if (AGG_OUT) AGG[(size_t)gr * HID + n] = v * dv2[r];
            }
        }
    }
}

// ---------------------------------------------------------------------------
// Helper kernels
// ---------------------------------------------------------------------------
__global__ void fill_f32(float* p, float v, int n) {
    int i = blockIdx.x * blockDim.x + threadIdx.x;
    if (i < n) p[i] = v;
}

// Convert nmat 128x128 f32 matrices to bf16, TRANSPOSED: Wt[m][n][k] = W[m][k][n]
__global__ void cvt_w_bf16_t(const float* __restrict__ W, __hip_bfloat16* __restrict__ Wt,
                             int nmat) {
    int i = blockIdx.x * blockDim.x + threadIdx.x;
    if (i >= nmat * HID * HID) return;
    int m = i >> 14;
    int k = (i >> 7) & 127;
    int n = i & 127;
    Wt[(size_t)m * HID * HID + (size_t)n * HID + k] = __float2bfloat16(W[i]);
}

__global__ void count_deg(const int* __restrict__ col, float* __restrict__ deg, int E) {
    int e = blockIdx.x * blockDim.x + threadIdx.x;
    if (e < E) atomicAdd(&deg[col[e]], 1.0f);
}

__global__ void compute_dinv(const float* __restrict__ deg, float* __restrict__ dinv, int n) {
    int i = blockIdx.x * blockDim.x + threadIdx.x;
    if (i < n) {
        float d = deg[i];
        dinv[i] = (d > 0.0f) ? rsqrtf(fmaxf(d, 1.0f)) : 0.0f;
    }
}

__global__ void compute_norm(const int* __restrict__ row, const int* __restrict__ col,
                             const float* __restrict__ dinv, float* __restrict__ norm, int E) {
    int e = blockIdx.x * blockDim.x + threadIdx.x;
    if (e < E) norm[e] = dinv[row[e]] * dinv[col[e]];
}

// Edge scatter: 16 threads per edge, 8 channels each (two 512B rows per wave)
__global__ void scatter_edges(const float* __restrict__ T, const int* __restrict__ row,
                              const int* __restrict__ col, const float* __restrict__ norm,
                              float* __restrict__ AGG, int E) {
    int tid = blockIdx.x * blockDim.x + threadIdx.x;
    int e = tid >> 4;
    if (e >= E) return;
    int g = (tid & 15) * 8;
    int r = row[e], c = col[e];
    float nv = norm[e];
    const float* src = T + (size_t)r * HID + g;
    float*       dst = AGG + (size_t)c * HID + g;
    float4 s0 = *(const float4*)(src);
    float4 s1 = *(const float4*)(src + 4);
    atomicAdd(dst + 0, s0.x * nv); atomicAdd(dst + 1, s0.y * nv);
    atomicAdd(dst + 2, s0.z * nv); atomicAdd(dst + 3, s0.w * nv);
    atomicAdd(dst + 4, s1.x * nv); atomicAdd(dst + 5, s1.y * nv);
    atomicAdd(dst + 6, s1.z * nv); atomicAdd(dst + 7, s1.w * nv);
}

__global__ void bias_relu(const float* __restrict__ AGG, const float* __restrict__ b,
                          float* __restrict__ H, int total) {
    int i = blockIdx.x * blockDim.x + threadIdx.x;
    if (i < total) H[i] = fmaxf(AGG[i] + b[i & 127], 0.0f);
}

// Pool accumulate: 16 threads per node, 8 channels each
__global__ void pool_acc(const float* __restrict__ H, const int* __restrict__ batch,
                         float* __restrict__ pooled, float* __restrict__ cnt, int n) {
    int tid = blockIdx.x * blockDim.x + threadIdx.x;
    int node = tid >> 4;
    if (node >= n) return;
    int g = (tid & 15) * 8;
    int b = batch[node];
    const float* src = H + (size_t)node * HID + g;
    float*       dst = pooled + (size_t)b * HID + g;
#pragma unroll
    for (int j = 0; j < 8; ++j) atomicAdd(dst + j, src[j]);
    if ((tid & 15) == 0) atomicAdd(&cnt[b], 1.0f);
}

__global__ void pool_div(float* __restrict__ pooled, const float* __restrict__ cnt, int total) {
    int i = blockIdx.x * blockDim.x + threadIdx.x;
    if (i < total) pooled[i] /= fmaxf(cnt[i >> 7], 1.0f);
}

// ---------------------------------------------------------------------------
// Orchestration
// ---------------------------------------------------------------------------
static inline size_t align_up(size_t x, size_t a) { return (x + a - 1) & ~(a - 1); }

extern "C" void kernel_launch(void* const* d_in, const int* in_sizes, int n_in,
                              void* d_out, int out_size, void* d_ws, size_t ws_size,
                              hipStream_t stream) {
    const float* x      = (const float*)d_in[0];
    const int*   edge   = (const int*)d_in[1];           // (2, E): rows then cols
    const int*   batch  = (const int*)d_in[2];
    const float* W_in   = (const float*)d_in[3];
    const float* b_in   = (const float*)d_in[4];
    const float* conv_W = (const float*)d_in[5];         // (4,128,128)
    const float* conv_b = (const float*)d_in[6];         // (4,128)
    const float* W_def  = (const float*)d_in[7];
    const float* b_def  = (const float*)d_in[8];
    const float* W_syn  = (const float*)d_in[9];
    const float* b_syn  = (const float*)d_in[10];
    const float* W_rel  = (const float*)d_in[11];
    const float* b_rel  = (const float*)d_in[12];
    float* out = (float*)d_out;

    const int N = N_NODES, E = N_EDGES;
    const int* erow = edge;
    const int* ecol = edge + E;

    // Carve workspace (node buffers padded to N_PAD rows so GEMM stores are unguarded)
    uintptr_t base = (uintptr_t)d_ws;
    size_t off = 0;
    auto carve = [&](size_t bytes) {
        uintptr_t p = base + off;
        off = align_up(off + bytes, 256);
        return p;
    };
    float* deg    = (float*)carve((size_t)N_PAD * 4);
    float* dinv   = (float*)carve((size_t)N_PAD * 4);
    float* norm   = (float*)carve((size_t)E * 4);
    float* H      = (float*)carve((size_t)N_PAD * HID * 4);
    float* T      = (float*)carve((size_t)N_PAD * HID * 4);
    float* AGG    = (float*)carve((size_t)N_PAD * HID * 4);
    float* pooled = (float*)carve((size_t)NUM_GRAPHS * HID * 4);
    float* cnt    = (float*)carve((size_t)NUM_GRAPHS * 4);
    // bf16 transposed weights: [0]=W_in, [1..4]=conv, [5]=def, [6]=syn, [7]=rel
    __hip_bfloat16* Wb = (__hip_bfloat16*)carve((size_t)8 * HID * HID * 2);
    (void)ws_size; (void)n_in; (void)in_sizes; (void)out_size;

    const int TPB = 256;
    auto blocks = [&](long long work) { return (unsigned)((work + TPB - 1) / TPB); };
    const int MSZ = HID * HID;

    // One-time weight prep: f32 -> bf16, transposed
    cvt_w_bf16_t<<<blocks(MSZ), TPB, 0, stream>>>(W_in,   Wb,            1);
    cvt_w_bf16_t<<<blocks(4 * MSZ), TPB, 0, stream>>>(conv_W, Wb + MSZ,  4);
    cvt_w_bf16_t<<<blocks(MSZ), TPB, 0, stream>>>(W_def,  Wb + 5 * MSZ,  1);
    cvt_w_bf16_t<<<blocks(MSZ), TPB, 0, stream>>>(W_syn,  Wb + 6 * MSZ,  1);
    cvt_w_bf16_t<<<blocks(MSZ), TPB, 0, stream>>>(W_rel,  Wb + 7 * MSZ,  1);

    // Degree (self-loop contributes 1), dinv, per-edge norm
    fill_f32<<<blocks(N_PAD), TPB, 0, stream>>>(deg, 1.0f, N_PAD);
    count_deg<<<blocks(E), TPB, 0, stream>>>(ecol, deg, E);
    compute_dinv<<<blocks(N_PAD), TPB, 0, stream>>>(deg, dinv, N_PAD);
    compute_norm<<<blocks(E), TPB, 0, stream>>>(erow, ecol, dinv, norm, E);

    const unsigned gemm_blocks = N_PAD / 128;   // 391

    // Input layer: H = relu(x @ W_in + b_in)   (guarded load for x, unguarded store)
    gemm128_wmma<true, true, false, false><<<gemm_blocks, TPB, 0, stream>>>(
        x, Wb, b_in, H, nullptr, nullptr, N, N_PAD);

    // 4 GCN layers: GEMM fused with AGG = T * dinv^2, then edge scatter, then bias+relu
    for (int l = 0; l < NUM_LAYERS; ++l) {
        gemm128_wmma<false, false, false, true><<<gemm_blocks, TPB, 0, stream>>>(
            H, Wb + (size_t)(1 + l) * MSZ, nullptr, T, AGG, dinv, N_PAD, N_PAD);
        scatter_edges<<<blocks((long long)E * 16), TPB, 0, stream>>>(T, erow, ecol, norm, AGG, E);
        bias_relu<<<blocks((long long)N * HID), TPB, 0, stream>>>(AGG, conv_b + l * HID, H, N * HID);
    }

    // Global mean pool
    fill_f32<<<blocks(NUM_GRAPHS * HID), TPB, 0, stream>>>(pooled, 0.0f, NUM_GRAPHS * HID);
    fill_f32<<<1, NUM_GRAPHS, 0, stream>>>(cnt, 0.0f, NUM_GRAPHS);
    pool_acc<<<blocks((long long)N * 16), TPB, 0, stream>>>(H, batch, pooled, cnt, N);
    pool_div<<<blocks(NUM_GRAPHS * HID), TPB, 0, stream>>>(pooled, cnt, NUM_GRAPHS * HID);

    // Heads via WMMA: (out_def, out_syn, out_rel) concatenated, M=64 each, guarded store
    const float* hb[3] = { b_def, b_syn, b_rel };
    for (int h = 0; h < 3; ++h) {
        gemm128_wmma<false, true, true, false><<<1, TPB, 0, stream>>>(
            pooled, Wb + (size_t)(5 + h) * MSZ, hb[h],
            out + (size_t)h * NUM_GRAPHS * HID, nullptr, nullptr,
            NUM_GRAPHS, NUM_GRAPHS);
    }
}
